// CoAtGFT_7962869367128
// MI455X (gfx1250) — matrix operand order repaired and checked by reading, not compiled, not run on hardware
//
#include <hip/hip_runtime.h>
#include <math.h>

typedef __attribute__((ext_vector_type(16))) _Float16 v16h;
typedef __attribute__((ext_vector_type(8)))  float    v8f;
typedef unsigned int v4u __attribute__((ext_vector_type(4)));
typedef int          v8i __attribute__((ext_vector_type(8)));
typedef int          v4i __attribute__((ext_vector_type(4)));

#define BLK   256
#define NTOK  49
#define CH    768
#define MIDC  48
#define SEL1  36
#define SEL2N 24

// LDS float counts
#define SMEM_FLOATS (NTOK*CH + CH + 64 + CH + CH + 64 + 64 + 64 + 64 + 16 + 8)

__device__ __forceinline__ float waveReduceSum(float v) {
#pragma unroll
  for (int off = 16; off >= 1; off >>= 1) v += __shfl_xor(v, off, 32);
  return v;
}

__device__ __forceinline__ float gelu_exact(float v) {
  return 0.5f * v * (1.0f + erff(v * 0.70710678118654752f));
}

__device__ __forceinline__ float sigmoidf(float v) {
  return 1.0f / (1.0f + expf(-v));
}

// One 16-wide N-tile of out[n] = sum_k vec[k] * W[k][n], via
// V_WMMA_F32_16X16X32_F16 with A = vec replicated in all 16 rows.
// vec: LDS f32 vector (length >= K); W: global row-major [K][ldw].
// A (contiguous k, as B): lanes0-15 halves -> K=0..15(+k0),
// lanes16-31 -> K=16..31(+k0). B: lanes0-15 halves -> K=0..15,
// lanes16-31 -> K=16..31, N = lane%16. D: M=0 row = VGPR0 of lanes 0-15.
__device__ __forceinline__ void wmma_matvec_tile(
    const float* vec, const float* __restrict__ W, int ldw, int K,
    int n0, float* out)
{
  const int lane = threadIdx.x & 31;
  const int grp  = lane >> 4;          // 0: lanes 0-15, 1: lanes 16-31
  const int nn   = n0 + (lane & 15);
  v8f acc = {0.f, 0.f, 0.f, 0.f, 0.f, 0.f, 0.f, 0.f};
  for (int k0 = 0; k0 < K; k0 += 32) {
    v16h a, b;
#pragma unroll
    for (int t = 0; t < 8; ++t) {
      int ka0 = k0 + grp * 16 + t;     // halves 0..7
      int ka1 = ka0 + 8;               // halves 8..15
      a[t]     = (_Float16)((ka0 < K) ? vec[ka0] : 0.f);
      a[t + 8] = (_Float16)((ka1 < K) ? vec[ka1] : 0.f);
    }
#pragma unroll
    for (int t = 0; t < 16; ++t) {
      int kb = k0 + grp * 16 + t;
      b[t] = (_Float16)((kb < K) ? W[kb * ldw + nn] : 0.f);
    }
    // (neg_a, A, neg_b, B, c_mod, C, reuse_a, reuse_b)
    acc = __builtin_amdgcn_wmma_f32_16x16x32_f16(false, a, false, b,
                                                 (short)0, acc, false, false);
  }
  if (lane < 16) out[nn] = acc[0];
}

__global__ void __launch_bounds__(BLK)
hier_se_fused(const float* __restrict__ x,
              const float* __restrict__ w11, const float* __restrict__ b11,
              const float* __restrict__ w12, const float* __restrict__ b12,
              const float* __restrict__ w21, const float* __restrict__ b21,
              const float* __restrict__ w22, const float* __restrict__ b22,
              const float* __restrict__ lng, const float* __restrict__ lnb,
              const float* __restrict__ clsw, const float* __restrict__ clsb,
              float* __restrict__ out)
{
  extern __shared__ __align__(16) float smem[];
  float* xs   = smem;              // 49*768  token tile, resident whole kernel
  float* vec  = xs + NTOK * CH;    // 768     s1 -> s2 -> pooled (reused)
  float* h    = vec + CH;          // 64      SE hidden
  float* g1   = h + 64;            // 768     stage-1 gates
  float* g2   = g1 + CH;           // 768     stage-2 gates (becomes g1*g2)
  float* ts   = g2 + CH;           // 64
  float* ts2  = ts + 64;           // 64
  float* sel  = ts2 + 64;          // 64      stage-1 keep mask
  float* sel2 = sel + 64;          // 64      stage-2 keep mask
  float* red  = sel2 + 64;         // 16      cross-wave reduce scratch
  float* st   = red + 16;          // 8       LN mu / inv-sigma

  const int tid  = threadIdx.x;
  const int lane = tid & 31;
  const int wv   = tid >> 5;
  const int b    = blockIdx.x;

  // ---- 1) stage x[b] (147 KB) into LDS: the ONLY HBM read of x ----
#if __has_builtin(__builtin_amdgcn_tensor_load_to_lds) && \
    __has_builtin(__builtin_amdgcn_s_wait_tensorcnt)
  if (wv == 0) {
    // Tensor Data Mover: 1-D flat copy, 37632 x f32, HBM -> LDS.
    // D# per ISA ch.8: group0 {count=1, lds_addr, global_addr, type=2},
    // group1 {data_size=4B, tensor_dim0=tile_dim0=37632, tensor_dim1=1,
    // tensor_dim0_stride=37632}; groups 2/3 zero (dims 2..4 unused).
    const unsigned long long ga =
        (unsigned long long)(const void*)(x + (size_t)b * (NTOK * CH));
    const unsigned int ldsBase = (unsigned int)__builtin_amdgcn_groupstaticsize();
    const unsigned int NE = (unsigned int)(NTOK * CH);       // 37632
    v4u g0 = { 1u,                                           // count=1
               ldsBase,                                      // lds_addr (bytes)
               (unsigned int)ga,                             // global_addr lo
               (unsigned int)((ga >> 32) & 0x01FFFFFFu) | (2u << 30) }; // hi|type=2
    v8i g1d = { (int)(2u << 16),                    // data_size = 4 bytes
                (int)((NE & 0xFFFFu) << 16),        // tensor_dim0 lo16
                (int)((NE >> 16) | (1u << 16)),     // tensor_dim0 hi16 | tensor_dim1=1
                (int)(NE << 16),                    // tile_dim0 = 37632 (16-bit)
                0,                                  // tile_dim1=0, tile_dim2=0
                (int)NE,                            // tensor_dim0_stride lo32
                0, 0 };
    v4i gz4 = { 0, 0, 0, 0 };
    v8i gz8 = { 0, 0, 0, 0, 0, 0, 0, 0 };
    // clang-23 / therock 6-arg form: (g0, g1, g2, g3, extra, cpol)
    __builtin_amdgcn_tensor_load_to_lds(g0, g1d, gz4, gz4, gz8, 0);
    __builtin_amdgcn_s_wait_tensorcnt((short)0);
  }
#else
  {
    const float4* xg = (const float4*)(x + (size_t)b * (NTOK * CH));
    float4* xs4 = (float4*)xs;
    for (int i = tid; i < NTOK * CH / 4; i += BLK) xs4[i] = xg[i];
  }
#endif
  __syncthreads();

  // ---- 2) s1 = mean over tokens ----
  for (int cc = tid; cc < CH; cc += BLK) {
    float s = 0.f;
    for (int n = 0; n < NTOK; ++n) s += xs[n * CH + cc];
    vec[cc] = s * (1.0f / NTOK);
  }
  __syncthreads();

  // ---- 3-6) stage-1 SE: gelu(s1@w1+b1) @ w2 + b2 -> sigmoid ----
  if (wv < 3) wmma_matvec_tile(vec, w11, MIDC, CH, wv * 16, h);
  __syncthreads();
  if (tid < MIDC) h[tid] = gelu_exact(h[tid] + b11[tid]);
  __syncthreads();
  for (int t = wv; t < CH / 16; t += 8) wmma_matvec_tile(h, w12, CH, MIDC, t * 16, g1);
  __syncthreads();
  for (int cc = tid; cc < CH; cc += BLK) g1[cc] = sigmoidf(g1[cc] + b12[cc]);
  __syncthreads();

  // ---- 7) per-token norms of gated tokens ----
  for (int n = wv; n < NTOK; n += 8) {
    float s = 0.f;
    for (int c = lane; c < CH; c += 32) { float v = xs[n * CH + c] * g1[c]; s += v * v; }
    s = waveReduceSum(s);
    if (lane == 0) ts[n] = sqrtf(s);
  }
  __syncthreads();

  // ---- 8) top-36: softmax(z-score(ts)) is monotonic in ts -> rank raw ts ----
  if (tid < NTOK) {
    float v = ts[tid];
    int r = 0;
    for (int m = 0; m < NTOK; ++m)
      r += (ts[m] > v) || (ts[m] == v && m < tid);   // lax.top_k tie order
    sel[tid] = (r < SEL1) ? 1.f : 0.f;
  }
  __syncthreads();

  // ---- 9) s2 = g1 * mean(selected x)  (gates are per-channel scalars) ----
  for (int cc = tid; cc < CH; cc += BLK) {
    float s = 0.f;
    for (int n = 0; n < NTOK; ++n) s += sel[n] * xs[n * CH + cc];
    vec[cc] = s * g1[cc] * (1.0f / SEL1);
  }
  __syncthreads();

  // ---- 10-13) stage-2 SE; fold g1 into g2 (combined per-channel gate) ----
  if (wv < 3) wmma_matvec_tile(vec, w21, MIDC, CH, wv * 16, h);
  __syncthreads();
  if (tid < MIDC) h[tid] = gelu_exact(h[tid] + b21[tid]);
  __syncthreads();
  for (int t = wv; t < CH / 16; t += 8) wmma_matvec_tile(h, w22, CH, MIDC, t * 16, g2);
  __syncthreads();
  for (int cc = tid; cc < CH; cc += BLK) g2[cc] = sigmoidf(g2[cc] + b22[cc]) * g1[cc];
  if (tid < 64) ts2[tid] = -1.f;
  __syncthreads();

  // ---- 14) norms of stage-2 gated tokens (selected only; wave-uniform skip) ----
  for (int n = wv; n < NTOK; n += 8) {
    if (sel[n] == 0.f) continue;
    float s = 0.f;
    for (int c = lane; c < CH; c += 32) { float v = xs[n * CH + c] * g2[c]; s += v * v; }
    s = waveReduceSum(s);
    if (lane == 0) ts2[n] = sqrtf(s);
  }
  __syncthreads();

  // ---- 15) top-24 among the 36 selected ----
  if (tid < NTOK) {
    float ok = 0.f;
    if (sel[tid] != 0.f) {
      float v = ts2[tid];
      int r = 0;
      for (int m = 0; m < NTOK; ++m)
        if (sel[m] != 0.f) r += (ts2[m] > v) || (ts2[m] == v && m < tid);
      ok = (r < SEL2N) ? 1.f : 0.f;
    }
    sel2[tid] = ok;
  }
  __syncthreads();

  // ---- 16) pooled = (g1*g2) * mean(sel2 x) ----
  for (int cc = tid; cc < CH; cc += BLK) {
    float s = 0.f;
    for (int n = 0; n < NTOK; ++n) s += sel2[n] * xs[n * CH + cc];
    vec[cc] = s * g2[cc] * (1.0f / SEL2N);
  }
  __syncthreads();

  // ---- 17) LayerNorm stats (biased var, per jnp.var) ----
  {
    float s = 0.f, q = 0.f;
    for (int cc = tid; cc < CH; cc += BLK) { float p = vec[cc]; s += p; q += p * p; }
    s = waveReduceSum(s); q = waveReduceSum(q);
    if (lane == 0) { red[wv] = s; red[8 + wv] = q; }
  }
  __syncthreads();
  if (tid == 0) {
    float S = 0.f, Q = 0.f;
    for (int w = 0; w < 8; ++w) { S += red[w]; Q += red[8 + w]; }
    float mu  = S * (1.0f / CH);
    float var = Q * (1.0f / CH) - mu * mu;
    st[0] = mu;
    st[1] = rsqrtf(var + 1e-5f);
  }
  __syncthreads();

  // ---- 18) classifier: wave j -> logit j ----
  if (wv < 5) {
    float mu = st[0], inv = st[1];
    float s = 0.f;
    for (int c = lane; c < CH; c += 32) {
      float nv = (vec[c] - mu) * inv * lng[c] + lnb[c];
      s += nv * clsw[c * 5 + wv];
    }
    s = waveReduceSum(s);
    if (lane == 0) out[(size_t)b * 5 + wv] = s + clsb[wv];
  }
}

extern "C" void kernel_launch(void* const* d_in, const int* in_sizes, int n_in,
                              void* d_out, int out_size, void* d_ws, size_t ws_size,
                              hipStream_t stream) {
  (void)n_in; (void)out_size; (void)d_ws; (void)ws_size;
  const float* x    = (const float*)d_in[0];
  const float* w11  = (const float*)d_in[1];
  const float* b11  = (const float*)d_in[2];
  const float* w12  = (const float*)d_in[3];
  const float* b12  = (const float*)d_in[4];
  const float* w21  = (const float*)d_in[5];
  const float* b21  = (const float*)d_in[6];
  const float* w22  = (const float*)d_in[7];
  const float* b22  = (const float*)d_in[8];
  const float* lng  = (const float*)d_in[9];
  const float* lnb  = (const float*)d_in[10];
  const float* clsw = (const float*)d_in[11];
  const float* clsb = (const float*)d_in[12];
  float* out = (float*)d_out;

  const int B = in_sizes[0] / (NTOK * CH);           // 2048
  const size_t smem = (size_t)SMEM_FLOATS * sizeof(float);  // 161,120 B -> 2 WGs/WGP
  (void)hipFuncSetAttribute((const void*)hier_se_fused,
                            hipFuncAttributeMaxDynamicSharedMemorySize, (int)smem);
  hier_se_fused<<<dim3(B), dim3(BLK), smem, stream>>>(
      x, w11, b11, w12, b12, w21, b21, w22, b22, lng, lnb, clsw, clsb, out);
}
